// BoldTokenizer_8254927143616
// MI455X (gfx1250) — matrix operand order repaired and checked by reading, not compiled, not run on hardware
//
#include <hip/hip_runtime.h>

typedef __attribute__((ext_vector_type(16))) _Float16 v16h;
typedef __attribute__((ext_vector_type(8)))  float    v8f;

#define DPATCH   256        // patch dim = K (16*16)
#define BROWS    128        // patch rows per workgroup (8 waves x 16)
#define WROWS    16         // patch rows per wave (one WMMA M tile)
#define VSIZE    4096       // vocab entries
#define NCHUNK   (VSIZE / 16)
#define NPATCH   16384      // 64 * 256 patches total

union F16Frag {
  uint4 u[2];
  v16h  v;
};

// ---------------------------------------------------------------------------
// Prep: split vocab fp32 -> f16 hi/lo (K-contiguous = B-fragment order) and
// precompute ||v||^2 per vocab row. One block per vocab row.
// ---------------------------------------------------------------------------
__global__ __launch_bounds__(256) void
vocab_prep(const float* __restrict__ vocab,
           unsigned short* __restrict__ vhi,
           unsigned short* __restrict__ vlo,
           float* __restrict__ v2tab) {
  __shared__ float sp[8];
  const int r = blockIdx.x;
  const int e = threadIdx.x;
  const float x = vocab[(size_t)r * DPATCH + e];
  const _Float16 h = (_Float16)x;
  const _Float16 l = (_Float16)(x - (float)h);      // f16x3 residual (~22 bits total)
  vhi[(size_t)r * DPATCH + e] = __builtin_bit_cast(unsigned short, h);
  vlo[(size_t)r * DPATCH + e] = __builtin_bit_cast(unsigned short, l);

  float p = x * x;
#pragma unroll
  for (int off = 16; off; off >>= 1) p += __shfl_xor(p, off, 32);
  if ((e & 31) == 0) sp[e >> 5] = p;
  __syncthreads();
  if (e == 0) {
    float t = 0.f;
#pragma unroll
    for (int w = 0; w < 8; ++w) t += sp[w];
    v2tab[r] = t;
  }
}

// ---------------------------------------------------------------------------
// Main: patchify + WMMA f16x3 nearest-vocab search.
// Block = 128 patch rows; wave w owns rows [w*16, w*16+16) and scans the whole
// vocab, so B operands are shared 8x through WGP$/L2 and the argmin is
// wave-local (no cross-wave reduction).
// ---------------------------------------------------------------------------
__global__ __launch_bounds__(256) void
bold_tokenizer_wmma(const float* __restrict__ images,
                    const unsigned short* __restrict__ vhi,
                    const unsigned short* __restrict__ vlo,
                    const float* __restrict__ v2tab,
                    float* __restrict__ out) {
  __shared__ unsigned short sAhi[BROWS * DPATCH];   // 64 KB
  __shared__ unsigned short sAlo[BROWS * DPATCH];   // 64 KB

  const int tid      = threadIdx.x;
  const int lane     = tid & 31;
  const int wave     = tid >> 5;
  const int row_base = blockIdx.x * BROWS;

  // ---- Phase 1: patchify gather + fp32 patches out + f16 hi/lo A tile to LDS
#pragma unroll
  for (int it = 0; it < 8; ++it) {
    const int idx = it * 256 + tid;      // (m, r) pair, 128 rows x 16 runs
    const int m   = idx & 127;           // consecutive tids -> consecutive rows
    const int r   = idx >> 7;            // row inside the 16x16 patch
    const int row = row_base + m;        // global patch index
    const int b   = row >> 8;
    const int n   = row & 255;
    const int pr  = n >> 4, pc = n & 15;
    const float* src = images + ((size_t)b << 16) + pr * 4096 + r * 256 + pc * 16;
    float* pdst = out + (size_t)row * DPATCH + r * 16;
    unsigned short* hi = sAhi + m * DPATCH + r * 16;
    unsigned short* lo = sAlo + m * DPATCH + r * 16;
#pragma unroll
    for (int q = 0; q < 4; ++q) {
      float4 x = ((const float4*)src)[q];
      ((float4*)pdst)[q] = x;                        // patches output (identity copy)
      float xs[4] = {x.x, x.y, x.z, x.w};
#pragma unroll
      for (int e = 0; e < 4; ++e) {
        _Float16 h = (_Float16)xs[e];
        _Float16 l = (_Float16)(xs[e] - (float)h);
        hi[q * 4 + e] = __builtin_bit_cast(unsigned short, h);
        lo[q * 4 + e] = __builtin_bit_cast(unsigned short, l);
      }
    }
  }
  __syncthreads();

  const int nsub = lane & 15;   // N (vocab col in chunk) == A row index within tile
  const int hi8  = lane >> 4;   // lane-half selector

  // ---- Hoist this wave's A fragments into registers (invariant over the scan).
  F16Frag ahiF[8], aloF[8];
#pragma unroll
  for (int t = 0; t < 8; ++t) {
    const int aidx = (wave * WROWS + nsub) * DPATCH + t * 32 + hi8 * 8;  // 16B aligned
    ahiF[t].u[0] = *(const uint4*)(sAhi + aidx);
    ahiF[t].u[1] = *(const uint4*)(sAhi + aidx + 16);
    aloF[t].u[0] = *(const uint4*)(sAlo + aidx);
    aloF[t].u[1] = *(const uint4*)(sAlo + aidx + 16);
  }

  float best[8];
  int   bidx[8];
#pragma unroll
  for (int j = 0; j < 8; ++j) { best[j] = __builtin_inff(); bidx[j] = 0; }

  for (int cc = 0; cc < NCHUNK; ++cc) {
    const int vcol = cc * 16 + nsub;                  // this lane's vocab column
    if (cc + 1 < NCHUNK) {
      __builtin_prefetch(vhi + (size_t)(vcol + 16) * DPATCH, 0, 1);
      __builtin_prefetch(vlo + (size_t)(vcol + 16) * DPATCH, 0, 1);
    }

    // lane covers K = kb + 8*hi8 + [0..7] and kb + 16 + 8*hi8 + [0..7]
    const uint4* ph = (const uint4*)(vhi + (size_t)vcol * DPATCH + hi8 * 8);
    const uint4* pl = (const uint4*)(vlo + (size_t)vcol * DPATCH + hi8 * 8);

    v8f C = {};
#pragma unroll
    for (int t = 0; t < 8; ++t) {                     // kb = 32*t
      F16Frag bhi, blo;
      bhi.u[0] = ph[t * 4];
      bhi.u[1] = ph[t * 4 + 2];
      blo.u[0] = pl[t * 4];
      blo.u[1] = pl[t * 4 + 2];

      // f16x3: hi*hi + hi*lo + lo*hi, fp32 accumulate
      C = __builtin_amdgcn_wmma_f32_16x16x32_f16(false, ahiF[t].v, false, bhi.v,
                                                 (short)0, C, false, false);
      C = __builtin_amdgcn_wmma_f32_16x16x32_f16(false, ahiF[t].v, false, blo.v,
                                                 (short)0, C, false, false);
      C = __builtin_amdgcn_wmma_f32_16x16x32_f16(false, aloF[t].v, false, bhi.v,
                                                 (short)0, C, false, false);
    }

    const float v2 = v2tab[vcol];
#pragma unroll
    for (int j = 0; j < 8; ++j) {                     // D VGPR j -> row = 8*hi8 + j
      float s = v2 - 2.0f * C[j];
      if (s < best[j]) { best[j] = s; bidx[j] = vcol; }  // strict <: lowest index wins
    }
  }

  // ---- Reduce across the 16 lanes of each half (same rows, different vocab cols)
#pragma unroll
  for (int j = 0; j < 8; ++j) {
    float bv = best[j]; int bi = bidx[j];
#pragma unroll
    for (int off = 8; off; off >>= 1) {               // masks < 16 stay within half
      float ov = __shfl_xor(bv, off, 32);
      int   oi = __shfl_xor(bi, off, 32);
      if (ov < bv || (ov == bv && oi < bi)) { bv = ov; bi = oi; }
    }
    if (nsub == 0) {
      const int row = row_base + wave * WROWS + hi8 * 8 + j;
      out[(size_t)NPATCH * DPATCH + row] = (float)bi;  // tokens after patches
    }
  }
}

extern "C" void kernel_launch(void* const* d_in, const int* in_sizes, int n_in,
                              void* d_out, int out_size, void* d_ws, size_t ws_size,
                              hipStream_t stream) {
  const float* images = (const float*)d_in[0];  // [64,256,256] fp32
  const float* vocab  = (const float*)d_in[1];  // [4096,256]  fp32
  float* out = (float*)d_out;                   // patches (16384*256) ++ tokens (16384)

  // workspace layout: vhi (2MB) | vlo (2MB) | v2tab (16KB)
  unsigned short* vhi   = (unsigned short*)d_ws;
  unsigned short* vlo   = vhi + (size_t)VSIZE * DPATCH;
  float*          v2tab = (float*)(vlo + (size_t)VSIZE * DPATCH);

  vocab_prep<<<VSIZE, 256, 0, stream>>>(vocab, vhi, vlo, v2tab);
  bold_tokenizer_wmma<<<NPATCH / BROWS, 256, 0, stream>>>(images, vhi, vlo, v2tab, out);
}
